// GAT_Model_18442589569745
// MI455X (gfx1250) — matrix operand order repaired and checked by reading, not compiled, not run on hardware
//
#include <hip/hip_runtime.h>
#include <hip/hip_bf16.h>

typedef __attribute__((ext_vector_type(16))) _Float16 v16h;
typedef __attribute__((ext_vector_type(8)))  _Float16 v8h;
typedef __attribute__((ext_vector_type(8)))  float    v8f;

#define WF   128   // feature width (= heads*hid = IN for this model)
#define EDIMC 16
#define NHEADS 4
#define NHID  32

// ---------------------------------------------------------------------------
// Utility kernels
// ---------------------------------------------------------------------------
__global__ void k_convert_f32_to_f16(const float* __restrict__ in,
                                     _Float16* __restrict__ out, long n) {
    long i = (long)blockIdx.x * blockDim.x + threadIdx.x;
    if (i < n) out[i] = (_Float16)in[i];
}

// W[k,col] (128x128 row-major) -> Wt16[col*128 + k] (f16, column-major)
__global__ void k_convert_w_t16(const float* __restrict__ W,
                                _Float16* __restrict__ Wt16) {
    int i = blockIdx.x * blockDim.x + threadIdx.x; // 0..16383
    int k = i >> 7, col = i & 127;
    Wt16[col * WF + k] = (_Float16)W[i];
}

// ---------------------------------------------------------------------------
// Self-loop attributes: loop_attr[n] = mean of edge_attr over incoming edges
// ---------------------------------------------------------------------------
__global__ void k_loop_attr_accum(const long long* __restrict__ ei,
                                  const float* __restrict__ ea,
                                  float* __restrict__ loop_attr,
                                  float* __restrict__ cnt, int E) {
    long i = (long)blockIdx.x * blockDim.x + threadIdx.x; // E*16 threads
    if (i >= (long)E * EDIMC) return;
    long e = i >> 4;
    int  t = (int)(i & 15);
    long dst = ei[E + e];
    atomicAdd(&loop_attr[dst * EDIMC + t], ea[i]);
    if (t == 0) atomicAdd(&cnt[dst], 1.0f);
}

__global__ void k_loop_attr_final(float* __restrict__ loop_attr,
                                  const float* __restrict__ cnt, int N) {
    long i = (long)blockIdx.x * blockDim.x + threadIdx.x; // N*16 threads
    if (i >= (long)N * EDIMC) return;
    loop_attr[i] /= fmaxf(cnt[i >> 4], 1.0f);
}

// ---------------------------------------------------------------------------
// Node GEMM: C[M x 128] = A16[M x 128] @ Bt16(col-major 128x128) + bias
// block = 256 threads = 8 waves; each wave: one 16x16 output tile (col tile).
// blockIdx.x selects 16-row tile (M % 16 == 0 guaranteed: N = 50000 = 16*3125)
// ---------------------------------------------------------------------------
__global__ void k_gemm_node(const _Float16* __restrict__ A,
                            const _Float16* __restrict__ Bt,
                            const float* __restrict__ bias,
                            float* __restrict__ C) {
    const int lane = threadIdx.x & 31;
    const int wave = threadIdx.x >> 5;      // 0..7 -> column tile
    const int r16  = lane & 15;
    const int hi   = lane >> 4;             // 0 or 1 (K-half select)
    const long row0 = (long)blockIdx.x * 16;
    const int  col  = wave * 16 + r16;      // global column 0..127

    const _Float16* arow = A  + (row0 + r16) * WF;
    const _Float16* bcol = Bt + (long)col * WF;

    v8f acc = {};
#pragma unroll
    for (int k0 = 0; k0 < WF; k0 += 32) {
        // A fragment (16x32 f16): lane holds row r16,
        //   elements 0..7  = K(k0 + hi*8 .. +7), 8..15 = K(k0+16+hi*8 .. +7)
        v8h alo = *(const v8h*)(arow + k0 + hi * 8);
        v8h ahi = *(const v8h*)(arow + k0 + 16 + hi * 8);
        v16h a;
#pragma unroll
        for (int i = 0; i < 8; ++i) { a[i] = alo[i]; a[i + 8] = ahi[i]; }
        // B fragment (32x16 f16): lane holds column r16,
        //   elements 0..7  = K(k0 + hi*8 .. +7), 8..15 = K(k0+16+hi*8 .. +7)
        v8h blo = *(const v8h*)(bcol + k0 + hi * 8);
        v8h bhi = *(const v8h*)(bcol + k0 + 16 + hi * 8);
        v16h b;
#pragma unroll
        for (int i = 0; i < 8; ++i) { b[i] = blo[i]; b[i + 8] = bhi[i]; }
        acc = __builtin_amdgcn_wmma_f32_16x16x32_f16(
            false, a, false, b, (short)0, acc, false, false);
    }
    const float bv = bias[col];
#pragma unroll
    for (int i = 0; i < 8; ++i) {
        // D layout: VGPR i -> M = i + 8*hi, N = r16
        C[(row0 + i + 8 * hi) * WF + col] = acc[i] + bv;
    }
}

// ---------------------------------------------------------------------------
// Edge score kernel: wave per edge.
//   m = xl[src] + xr[dst] + ea@We (fused, We in LDS)
//   e_h = sum_c leakyrelu(m,0.2)*att[h,c];  atomicMax segment max (uint map)
// ---------------------------------------------------------------------------
__device__ __forceinline__ unsigned fmap(float f) {
    unsigned u = __float_as_uint(f);
    return (f >= 0.0f) ? (u | 0x80000000u) : ~u;
}
__device__ __forceinline__ float funmap(unsigned u) {
    return (u & 0x80000000u) ? __uint_as_float(u & 0x7FFFFFFFu)
                             : __uint_as_float(~u);
}

__global__ void k_edge_score(const long long* __restrict__ ei,
                             const float* __restrict__ ea,
                             const float* __restrict__ loop_attr,
                             const float* __restrict__ xl,
                             const float* __restrict__ xr,
                             const float* __restrict__ We,   // 16 x 128
                             const float* __restrict__ att,  // 4 x 32
                             float* __restrict__ escore,     // E' x 4
                             unsigned* __restrict__ emax,    // N x 4 (mapped)
                             int E, int N) {
    __shared__ float sWe[EDIMC * WF];
    __shared__ float sAtt[WF];
    for (int i = threadIdx.x; i < EDIMC * WF; i += blockDim.x) sWe[i] = We[i];
    if (threadIdx.x < WF) sAtt[threadIdx.x] = att[threadIdx.x];
    __syncthreads();

    const int lane = threadIdx.x & 31;
    const long Eaug = (long)E + N;
    long e = (long)blockIdx.x * (blockDim.x >> 5) + (threadIdx.x >> 5);
    if (e >= Eaug) return;

    long src, dst; const float* earow;
    if (e < E) { src = ei[e]; dst = ei[E + e]; earow = ea + e * EDIMC; }
    else       { src = e - E; dst = src;       earow = loop_attr + src * EDIMC; }

    const int j = lane * 4;       // 4 consecutive feature columns per lane
    const int h = lane >> 3;      // head = j / 32

    float m[4];
#pragma unroll
    for (int c = 0; c < 4; ++c)
        m[c] = xl[src * WF + j + c] + xr[dst * WF + j + c];
#pragma unroll
    for (int t = 0; t < EDIMC; ++t) {
        const float eat = earow[t];
#pragma unroll
        for (int c = 0; c < 4; ++c) m[c] += eat * sWe[t * WF + j + c];
    }
    float partial = 0.0f;
#pragma unroll
    for (int c = 0; c < 4; ++c) {
        float v = m[c];
        v = (v > 0.0f) ? v : 0.2f * v;                  // leaky_relu(0.2)
        partial += v * sAtt[h * NHID + (lane & 7) * 4 + c];
    }
    // reduce over the 8 lanes of this head
    partial += __shfl_xor(partial, 1, 32);
    partial += __shfl_xor(partial, 2, 32);
    partial += __shfl_xor(partial, 4, 32);
    if ((lane & 7) == 0) {
        escore[e * NHEADS + h] = partial;
        atomicMax(&emax[dst * NHEADS + h], fmap(partial));
    }
}

// ---------------------------------------------------------------------------
// exp(e - max) and denominator accumulation. One thread per (edge, head).
// Overwrites escore with ex in place.
// ---------------------------------------------------------------------------
__global__ void k_edge_exp(const long long* __restrict__ ei,
                           float* __restrict__ escore,
                           const unsigned* __restrict__ emax,
                           float* __restrict__ denom, int E, int N) {
    long i = (long)blockIdx.x * blockDim.x + threadIdx.x;
    long Eaug = (long)E + N;
    if (i >= Eaug * NHEADS) return;
    long e = i >> 2;
    int  h = (int)(i & 3);
    long dst = (e < E) ? ei[E + e] : (e - E);
    float mx = funmap(emax[dst * NHEADS + h]);
    float ex = expf(escore[i] - mx);
    escore[i] = ex;
    atomicAdd(&denom[dst * NHEADS + h], ex);
}

// ---------------------------------------------------------------------------
// Aggregation: out[dst] += alpha * xl[src]. Wave per edge, float atomics.
// ---------------------------------------------------------------------------
__global__ void k_edge_aggregate(const long long* __restrict__ ei,
                                 const float* __restrict__ exbuf,
                                 const float* __restrict__ denom,
                                 const float* __restrict__ xl,
                                 float* __restrict__ outacc, int E, int N) {
    const int lane = threadIdx.x & 31;
    const long Eaug = (long)E + N;
    long e = (long)blockIdx.x * (blockDim.x >> 5) + (threadIdx.x >> 5);
    if (e >= Eaug) return;
    long src, dst;
    if (e < E) { src = ei[e]; dst = ei[E + e]; }
    else       { src = e - E; dst = src; }
    const int j = lane * 4;
    const int h = lane >> 3;
    float alpha = exbuf[e * NHEADS + h] /
                  (denom[dst * NHEADS + h] + 1e-16f);
#pragma unroll
    for (int c = 0; c < 4; ++c)
        atomicAdd(&outacc[dst * WF + j + c], alpha * xl[src * WF + j + c]);
}

// ---------------------------------------------------------------------------
// Finalize: add bias, exact GELU, write f32 h and f16 copy for next GEMM.
// ---------------------------------------------------------------------------
__global__ void k_node_finalize(const float* __restrict__ outacc,
                                const float* __restrict__ b,
                                float* __restrict__ hbuf,
                                _Float16* __restrict__ h16, long n) {
    long i = (long)blockIdx.x * blockDim.x + threadIdx.x;
    if (i >= n) return;
    float v = outacc[i] + b[i & (WF - 1)];
    float g = 0.5f * v * (1.0f + erff(v * 0.70710678118654752f));
    hbuf[i] = g;
    h16[i] = (_Float16)g;
}

// ---------------------------------------------------------------------------
// Classifier: logit[n] = h[n,:] . Wc + bc   (wave per node)
// ---------------------------------------------------------------------------
__global__ void k_classifier(const float* __restrict__ h,
                             const float* __restrict__ Wc,
                             const float* __restrict__ bc,
                             float* __restrict__ out, int N) {
    const int lane = threadIdx.x & 31;
    long n = (long)blockIdx.x * (blockDim.x >> 5) + (threadIdx.x >> 5);
    if (n >= N) return;
    float s = 0.0f;
#pragma unroll
    for (int c = 0; c < 4; ++c)
        s += h[n * WF + lane * 4 + c] * Wc[lane * 4 + c];
    for (int m = 1; m < 32; m <<= 1) s += __shfl_xor(s, m, 32);
    if (lane == 0) out[n] = s + bc[0];
}

// ---------------------------------------------------------------------------
// Orchestration
// ---------------------------------------------------------------------------
extern "C" void kernel_launch(void* const* d_in, const int* in_sizes, int n_in,
                              void* d_out, int out_size, void* d_ws, size_t ws_size,
                              hipStream_t stream) {
    const float*     x    = (const float*)d_in[0];
    const long long* ei   = (const long long*)d_in[1];  // int64 (2,E)
    const float*     ea   = (const float*)d_in[2];
    const float*     Wl[2]  = { (const float*)d_in[3],  (const float*)d_in[10] };
    const float*     bl[2]  = { (const float*)d_in[4],  (const float*)d_in[11] };
    const float*     Wr[2]  = { (const float*)d_in[5],  (const float*)d_in[12] };
    const float*     br[2]  = { (const float*)d_in[6],  (const float*)d_in[13] };
    const float*     Wem[2] = { (const float*)d_in[7],  (const float*)d_in[14] };
    const float*     att[2] = { (const float*)d_in[8],  (const float*)d_in[15] };
    const float*     bb[2]  = { (const float*)d_in[9],  (const float*)d_in[16] };
    const float*     Wc   = (const float*)d_in[17];
    const float*     bc   = (const float*)d_in[18];
    float*           out  = (float*)d_out;

    const int N = in_sizes[0] / WF;
    const int E = in_sizes[2] / EDIMC;
    const long Eaug = (long)E + N;

    // ---- workspace layout ----
    size_t off = 0;
    auto alloc = [&](size_t bytes) -> char* {
        off = (off + 255) & ~(size_t)255;
        char* p = (char*)d_ws + off;
        off += bytes;
        return p;
    };
    _Float16* in16   = (_Float16*)alloc((size_t)N * WF * sizeof(_Float16));
    _Float16* wtl16  = (_Float16*)alloc((size_t)WF * WF * sizeof(_Float16));
    _Float16* wtr16  = (_Float16*)alloc((size_t)WF * WF * sizeof(_Float16));
    float*    xl     = (float*)alloc((size_t)N * WF * sizeof(float));
    float*    xr     = (float*)alloc((size_t)N * WF * sizeof(float));
    float*    outacc = (float*)alloc((size_t)N * WF * sizeof(float));
    float*    hbuf   = (float*)alloc((size_t)N * WF * sizeof(float));
    float*    lattr  = (float*)alloc((size_t)N * EDIMC * sizeof(float));
    float*    cnt    = (float*)alloc((size_t)N * sizeof(float));
    float*    escore = (float*)alloc((size_t)Eaug * NHEADS * sizeof(float));
    unsigned* emax   = (unsigned*)alloc((size_t)N * NHEADS * sizeof(unsigned));
    float*    denom  = (float*)alloc((size_t)N * NHEADS * sizeof(float));
    (void)ws_size;

    const int TB = 256;
    const int wavesPerBlock = TB / 32;

    // ---- self-loop attributes (mean of incoming edge_attr) ----
    hipMemsetAsync(lattr, 0, (size_t)N * EDIMC * sizeof(float), stream);
    hipMemsetAsync(cnt,   0, (size_t)N * sizeof(float), stream);
    {
        long tot = (long)E * EDIMC;
        k_loop_attr_accum<<<(int)((tot + TB - 1) / TB), TB, 0, stream>>>(
            ei, ea, lattr, cnt, E);
        long tot2 = (long)N * EDIMC;
        k_loop_attr_final<<<(int)((tot2 + TB - 1) / TB), TB, 0, stream>>>(
            lattr, cnt, N);
    }

    // ---- input -> f16 ----
    {
        long tot = (long)N * WF;
        k_convert_f32_to_f16<<<(int)((tot + TB - 1) / TB), TB, 0, stream>>>(
            x, in16, tot);
    }

    // ---- two GATv2 layers ----
    for (int L = 0; L < 2; ++L) {
        // weights -> transposed f16
        k_convert_w_t16<<<(WF * WF) / TB, TB, 0, stream>>>(Wl[L], wtl16);
        k_convert_w_t16<<<(WF * WF) / TB, TB, 0, stream>>>(Wr[L], wtr16);
        // node transforms (WMMA)
        k_gemm_node<<<N / 16, TB, 0, stream>>>(in16, wtl16, bl[L], xl);
        k_gemm_node<<<N / 16, TB, 0, stream>>>(in16, wtr16, br[L], xr);
        // segment softmax state
        hipMemsetAsync(emax,   0, (size_t)N * NHEADS * sizeof(unsigned), stream);
        hipMemsetAsync(denom,  0, (size_t)N * NHEADS * sizeof(float), stream);
        hipMemsetAsync(outacc, 0, (size_t)N * WF * sizeof(float), stream);
        // edge scores + segment max
        k_edge_score<<<(int)((Eaug + wavesPerBlock - 1) / wavesPerBlock), TB, 0,
                       stream>>>(ei, ea, lattr, xl, xr, Wem[L], att[L],
                                 escore, emax, E, N);
        // exp + denominator
        {
            long tot = Eaug * NHEADS;
            k_edge_exp<<<(int)((tot + TB - 1) / TB), TB, 0, stream>>>(
                ei, escore, emax, denom, E, N);
        }
        // weighted aggregation
        k_edge_aggregate<<<(int)((Eaug + wavesPerBlock - 1) / wavesPerBlock), TB,
                           0, stream>>>(ei, escore, denom, xl, outacc, E, N);
        // bias + gelu; produce next-layer f16 input
        {
            long tot = (long)N * WF;
            k_node_finalize<<<(int)((tot + TB - 1) / TB), TB, 0, stream>>>(
                outacc, bb[L], hbuf, in16, tot);
        }
    }

    // ---- classifier ----
    k_classifier<<<(int)((N + wavesPerBlock - 1) / wavesPerBlock), TB, 0,
                   stream>>>(hbuf, Wc, bc, out, N);
}
